// Decoder_72670846648424
// MI455X (gfx1250) — matrix operand order repaired and checked by reading, not compiled, not run on hardware
//
#include <hip/hip_runtime.h>

// ---------------------------------------------------------------------------
// Types
// ---------------------------------------------------------------------------
typedef __bf16 bf16;
typedef __attribute__((ext_vector_type(16))) __bf16 v16bf;
typedef __attribute__((ext_vector_type(8)))  float  v8f;

#define Bsz    64
#define Tlen   1024
#define Lsteps 500
#define Edim   128
#define Odim   80
#define Pdim   512
#define IDIM   640
#define RESIN  1280
#define NWG    64
#define NTHR   256

__device__ __forceinline__ float sigmoidf_(float x) { return 1.0f / (1.0f + __expf(-x)); }

__device__ __forceinline__ v8f zero8() {
  v8f c;
#pragma unroll
  for (int i = 0; i < 8; ++i) c[i] = 0.0f;
  return c;
}

// ---------------------------------------------------------------------------
// WMMA fragment loaders (CDNA5 16x16x32 bf16 layouts, ISA 7.12.2)
//   A (16x32): lane L holds row M=L%16; elems 0..7 -> K = base..base+7,
//              elems 8..15 -> K = 16+base..16+base+7, base = (L<16?0:8)
//   B (32x16): lane L holds col N=L%16; elem e -> K = base + (e&7) + 16*(e>>3), same as A
//   C/D:       lane L, vgpr v -> row = v + (L<16?0:8), col = L%16
// ---------------------------------------------------------------------------
__device__ __forceinline__ v16bf load_a_frag(const float* A, int lda, int m0, int k0, int Klim) {
  int lane = threadIdx.x & 31;
  const float* row = A + (size_t)(m0 + (lane & 15)) * lda;
  int off = (lane >> 4) << 3;        // 0 or 8
  int b0 = k0 + off;
  int b1 = k0 + 16 + off;
  v16bf a;
  if (b1 + 7 < Klim) {
    v8f x0 = *(const v8f*)(row + b0);
    v8f x1 = *(const v8f*)(row + b1);
#pragma unroll
    for (int i = 0; i < 8; ++i) { a[i] = (__bf16)x0[i]; a[8 + i] = (__bf16)x1[i]; }
  } else {
#pragma unroll
    for (int i = 0; i < 8; ++i) {
      int ka = b0 + i, kb = b1 + i;
      a[i]     = (__bf16)((ka < Klim) ? row[ka] : 0.0f);
      a[8 + i] = (__bf16)((kb < Klim) ? row[kb] : 0.0f);
    }
  }
  return a;
}

__device__ __forceinline__ v16bf load_b_frag(const unsigned short* pack, int ktiles, int nt, int kt) {
  int lane = threadIdx.x & 31;
  const bf16* p = (const bf16*)pack + (((size_t)(nt * ktiles + kt)) * 32 + lane) * 16;
  return *(const v16bf*)p;           // 32B aligned contiguous fragment
}

__device__ __forceinline__ v8f gemm_tile(v8f c, const float* A, int lda, int m0, int Klim,
                                         int ktiles, const unsigned short* pack, int nt) {
  for (int kt = 0; kt < ktiles; ++kt) {
    v16bf a = load_a_frag(A, lda, m0, kt * 32, Klim);
    v16bf b = load_b_frag(pack, ktiles, nt, kt);
    c = __builtin_amdgcn_wmma_f32_16x16x32_bf16(false, a, false, b, (short)0, c, false, false);
  }
  return c;
}

// ---------------------------------------------------------------------------
// Prep kernels
// ---------------------------------------------------------------------------
// Pack row-major W (N x K) into B-fragment layout: y = x @ W^T  => B[k][n] = W[n][k]
__global__ void pack_w_kernel(const float* __restrict__ W, unsigned short* __restrict__ dst,
                              int N, int K, int ktiles, long total) {
  long id = (long)blockIdx.x * blockDim.x + threadIdx.x;
  if (id >= total) return;
  int e    = (int)(id & 15);
  int lane = (int)((id >> 4) & 31);
  long tile = id >> 9;
  int kt = (int)(tile % ktiles);
  int nt = (int)(tile / ktiles);
  int n = nt * 16 + (lane & 15);
  int k = kt * 32 + ((lane >> 4) << 3) + (e & 7) + ((e >> 3) << 4);
  float v = (n < N && k < K) ? W[(size_t)n * K + k] : 0.0f;
  ((bf16*)dst)[id] = (bf16)v;
}

__global__ void f32_to_bf16_kernel(const float* __restrict__ src, unsigned short* __restrict__ dst, long n) {
  long i = (long)blockIdx.x * blockDim.x + threadIdx.x;
  if (i < n) ((bf16*)dst)[i] = (bf16)src[i];
}

// Fold stop head through projection: q[h] = sum_o stopw[o]*projw[o,h]; q[512] = stopw.projb + stopb
__global__ void fuse_stop_kernel(const float* __restrict__ projw, const float* __restrict__ projb,
                                 const float* __restrict__ stopw, const float* __restrict__ stopb,
                                 float* __restrict__ q) {
  int h = threadIdx.x;
  if (h < 512) {
    float acc = 0.0f;
    for (int o = 0; o < 80; ++o) acc += stopw[o] * projw[(size_t)o * 512 + h];
    q[h] = acc;
  }
  if (h == 0) {
    float acc = stopb[0];
    for (int o = 0; o < 80; ++o) acc += stopw[o] * projb[o];
    q[512] = acc;
  }
}

// Fused prenet: per 16-row tile, relu(ys@W1^T+b1) kept in LDS, then relu(@W2^T+b2) -> bf16
__global__ __launch_bounds__(NTHR) void prenet_kernel(const float* __restrict__ ys,
                                                      const unsigned short* __restrict__ pk1,
                                                      const float* __restrict__ b1,
                                                      const unsigned short* __restrict__ pk2,
                                                      const float* __restrict__ b2,
                                                      unsigned short* __restrict__ pre_all) {
  __shared__ __align__(32) float s_hid[16 * 512];   // 32KB
  int r0   = blockIdx.x * 16;
  int wave = threadIdx.x >> 5;
  int lane = threadIdx.x & 31;
  for (int nt = wave; nt < 32; nt += 8) {
    v8f c = gemm_tile(zero8(), ys, 80, r0, 80, 3, pk1, nt);
    int n = nt * 16 + (lane & 15);
    float bv = b1[n];
    int mb = (lane >> 4) << 3;
#pragma unroll
    for (int v = 0; v < 8; ++v) { float x = c[v] + bv; s_hid[(size_t)(mb + v) * 512 + n] = fmaxf(x, 0.0f); }
  }
  __syncthreads();
  for (int nt = wave; nt < 32; nt += 8) {
    v8f c = gemm_tile(zero8(), s_hid, 512, 0, 512, 16, pk2, nt);
    int n = nt * 16 + (lane & 15);
    float bv = b2[n];
    int mb = (lane >> 4) << 3;
#pragma unroll
    for (int v = 0; v < 8; ++v) {
      float x = fmaxf(c[v] + bv, 0.0f);
      ((bf16*)pre_all)[(size_t)(r0 + mb + v) * 512 + n] = (bf16)x;
    }
  }
}

// ---------------------------------------------------------------------------
// Persistent decode kernel
// ---------------------------------------------------------------------------
struct DecArgs {
  const unsigned short* pk_gru_wih[4];
  const unsigned short* pk_gru_whh[4];
  const unsigned short* pk_lstm_wih[2];
  const unsigned short* pk_lstm_whh[2];
  const unsigned short* pk_proj;
  const float* gru_bih[4]; const float* gru_bhh[4];
  const float* mol_w1[2]; const float* mol_b1[2]; const float* mol_w2[2]; const float* mol_b2[2];
  const float* lstm_bih[2]; const float* lstm_bhh[2];
  const float* proj_b;
  const float* q;                    // 513 floats: q[0..511], qb at [512]
  const unsigned short* enc_bf[2];   // [0]=enc_p (acoustic branch), [1]=enc_z (prosody)
  const unsigned short* pre_all;     // (B*L, 512) bf16
  float *a_in, *p_in;                // 64x640 = [prenet | ctx]
  float *h_gru;                      // 4 x 64 x 256
  float *res_in;                     // 64x1280 = [ctx_a | ha_f | ha_b | ctx_p | hp_f | hp_b]
  float *hl, *cl;                    // 2 x 64 x 256
  float *res_out;                    // 64x512
  float *gi, *gh;                    // 4 x 64 x 768
  float *g_lstm;                     // 2 x 64 x 1024
  unsigned *bar, *gen;
  float *outs, *logits;
};

__device__ __forceinline__ void gsync(unsigned* bar, unsigned* gen) {
  __syncthreads();
  __threadfence();
  if (threadIdx.x == 0) {
    volatile unsigned* vg = (volatile unsigned*)gen;
    unsigned g = *vg;
    if (atomicAdd(bar, 1u) == NWG - 1u) {
      *(volatile unsigned*)bar = 0u;
      __threadfence();
      atomicAdd(gen, 1u);
    } else {
      while (*vg == g) __builtin_amdgcn_s_sleep(2);
    }
    __threadfence();
  }
  __syncthreads();
}

__global__ __launch_bounds__(NTHR) void decode_kernel(DecArgs A) {
  __shared__ float s_rnn[512];
  __shared__ float s_hid[256];
  __shared__ float s_raw[16];
  __shared__ float s_probs[1024];

  const int blk  = blockIdx.x;
  const int tid  = threadIdx.x;
  const int lane = tid & 31;
  const int wid  = blk * 8 + (tid >> 5);      // 512 waves

  for (int t = 0; t < Lsteps; ++t) {
    // ---- P1: GRU GEMMs: gi = a/p_in @ wih^T + bih ; gh = h @ whh^T + bhh
    for (int id = wid; id < 1536; id += NWG * 8) {
      int gru = id / 384;
      int rem = id % 384;
      int mat = rem / 192; rem %= 192;
      int mt = rem / 48, nt = rem % 48;
      const float* Ain; int lda, Kd, ktiles; const unsigned short* pk; const float* bias; float* out;
      if (mat == 0) {
        Ain = (gru < 2) ? A.a_in : A.p_in; lda = IDIM; Kd = IDIM; ktiles = 20;
        pk = A.pk_gru_wih[gru]; bias = A.gru_bih[gru]; out = A.gi + gru * 49152;
      } else {
        Ain = A.h_gru + gru * 16384; lda = 256; Kd = 256; ktiles = 8;
        pk = A.pk_gru_whh[gru]; bias = A.gru_bhh[gru]; out = A.gh + gru * 49152;
      }
      v8f c = gemm_tile(zero8(), Ain, lda, mt * 16, Kd, ktiles, pk, nt);
      int n = nt * 16 + (lane & 15);
      float bv = bias[n];
      int mbase = mt * 16 + ((lane >> 4) << 3);
#pragma unroll
      for (int v = 0; v < 8; ++v) out[(size_t)(mbase + v) * 768 + n] = c[v] + bv;
    }
    gsync(A.bar, A.gen);

    // ---- P2: GRU cell update; write h and rnn columns of res_in
    for (int i = blk * NTHR + tid; i < 4 * 64 * 256; i += NWG * NTHR) {
      int gru = i >> 14; int b = (i >> 8) & 63; int j = i & 255;
      const float* gi = A.gi + gru * 49152 + b * 768;
      const float* gh = A.gh + gru * 49152 + b * 768;
      float r  = sigmoidf_(gi[j] + gh[j]);
      float z  = sigmoidf_(gi[256 + j] + gh[256 + j]);
      float nn = tanhf(gi[512 + j] + r * gh[512 + j]);
      float* hp = A.h_gru + gru * 16384 + b * 256 + j;
      float hnew = (1.0f - z) * nn + z * (*hp);
      *hp = hnew;
      int col = 128 + (gru & 1) * 256 + (gru >> 1) * 640 + j;  // 128/384/768/1024
      A.res_in[(size_t)b * RESIN + col] = hnew;
    }
    gsync(A.bar, A.gen);

    // ---- P4: MOL attention + head (per (branch,b) pair; 2 pairs per block)
    for (int pr = blk; pr < 128; pr += NWG) {
      int branch = pr >> 6, b = pr & 63;
      for (int i = tid; i < 512; i += NTHR)
        s_rnn[i] = A.res_in[(size_t)b * RESIN + (branch ? 768 : 128) + i];
      __syncthreads();
      { // hid = tanh(W1 @ rnn + b1), 256 outputs over 256 threads
        const float* w = A.mol_w1[branch] + (size_t)tid * 512;
        float acc = A.mol_b1[branch][tid];
        for (int k = 0; k < 512; ++k) acc += w[k] * s_rnn[k];
        s_hid[tid] = tanhf(acc);
      }
      __syncthreads();
      if (tid < 15) {
        const float* w = A.mol_w2[branch] + (size_t)tid * 256;
        float acc = A.mol_b2[branch][tid];
        for (int k = 0; k < 256; ++k) acc += w[k] * s_hid[k];
        s_raw[tid] = acc;
      }
      __syncthreads();
      float wk[5], mu[5], sinv[5];
      {
        float mx = -1e30f;
        for (int k = 0; k < 5; ++k) mx = fmaxf(mx, s_raw[k]);
        float sm = 0.0f;
        for (int k = 0; k < 5; ++k) { wk[k] = __expf(s_raw[k] - mx); sm += wk[k]; }
        float rs = 1.0f / sm;
        for (int k = 0; k < 5; ++k) wk[k] *= rs;
        for (int k = 0; k < 5; ++k) { mu[k] = s_raw[5 + k]; sinv[k] = __expf(-s_raw[10 + k]); }
      }
      for (int tt = tid; tt < Tlen; tt += NTHR) {
        float x = (float)tt, p = 0.0f;
#pragma unroll
        for (int k = 0; k < 5; ++k)
          p += wk[k] * (sigmoidf_((x + 0.5f - mu[k]) * sinv[k]) -
                        sigmoidf_((x - 0.5f - mu[k]) * sinv[k]));
        s_probs[tt] = p;
      }
      __syncthreads();
      if (tid < Edim) { // ctx[e] = sum_t probs[t] * enc[b,t,e]   (bf16 encoder, L2 resident)
        const bf16* enc = (const bf16*)A.enc_bf[branch] + (size_t)b * Tlen * Edim + tid;
        float acc = 0.0f;
        for (int tt = 0; tt < Tlen; ++tt) acc += s_probs[tt] * (float)enc[(size_t)tt * Edim];
        A.res_in[(size_t)b * RESIN + (branch ? 640 : 0) + tid] = acc;
        float* dst = branch ? A.p_in : A.a_in;
        dst[(size_t)b * IDIM + 512 + tid] = acc;
      }
      __syncthreads();
    }
    gsync(A.bar, A.gen);

    // ---- P6: LSTM GEMMs: g = res_in@wih^T + h@whh^T + bih + bhh (512 tiles, 1/wave)
    {
      int id = wid;
      int l = id >> 8; int rem = id & 255; int mt = rem >> 6; int nt = rem & 63;
      v8f c = gemm_tile(zero8(), A.res_in, RESIN, mt * 16, RESIN, 40, A.pk_lstm_wih[l], nt);
      c = gemm_tile(c, A.hl + l * 16384, 256, mt * 16, 256, 8, A.pk_lstm_whh[l], nt);
      int n = nt * 16 + (lane & 15);
      float bv = A.lstm_bih[l][n] + A.lstm_bhh[l][n];
      int mbase = mt * 16 + ((lane >> 4) << 3);
#pragma unroll
      for (int v = 0; v < 8; ++v)
        A.g_lstm[(size_t)l * 65536 + (size_t)(mbase + v) * 1024 + n] = c[v] + bv;
    }
    gsync(A.bar, A.gen);

    // ---- P7: LSTM cell + zoneout (eval mixing) -> hl/cl and res_out
    for (int i = blk * NTHR + tid; i < 2 * 64 * 256; i += NWG * NTHR) {
      int l = i >> 14; int b = (i >> 8) & 63; int j = i & 255;
      const float* g = A.g_lstm + (size_t)l * 65536 + (size_t)b * 1024;
      float ii = sigmoidf_(g[j]);
      float ff = sigmoidf_(g[256 + j]);
      float gg = tanhf(g[512 + j]);
      float oo = sigmoidf_(g[768 + j]);
      float* cp = A.cl + l * 16384 + b * 256 + j;
      float* hp = A.hl + l * 16384 + b * 256 + j;
      float c2 = ff * (*cp) + ii * gg;
      float h2 = oo * tanhf(c2);
      float hn = 0.1f * (*hp) + 0.9f * h2;
      float cn = 0.1f * (*cp) + 0.9f * c2;
      *hp = hn; *cp = cn;
      A.res_out[(size_t)b * 512 + l * 256 + j] = hn;
    }
    gsync(A.bar, A.gen);

    // ---- P8: projection (WMMA) -> outs; fused stop logits; prefetch next pre row
    if (wid < 20) {
      int mt = wid / 5, nt = wid % 5;
      v8f c = gemm_tile(zero8(), A.res_out, 512, mt * 16, 512, 16, A.pk_proj, nt);
      int n = nt * 16 + (lane & 15);
      float bv = A.proj_b[n];
      int mbase = mt * 16 + ((lane >> 4) << 3);
#pragma unroll
      for (int v = 0; v < 8; ++v) {
        int b = mbase + v;
        A.outs[((size_t)b * Lsteps + t) * Odim + n] = c[v] + bv;
      }
    } else if (wid < 22) {
      int b = (wid - 20) * 32 + lane;
      float acc = A.q[512];
      const float* r = A.res_out + (size_t)b * 512;
      for (int h = 0; h < 512; ++h) acc += r[h] * A.q[h];
      A.logits[(size_t)b * Lsteps + t] = acc;
    }
    if (blk >= 32) { // copy prenet(ys[:,t]) -> a_in/p_in[:, :512] for step t+1
      for (int i = (blk - 32) * NTHR + tid; i < 64 * 512; i += 32 * NTHR) {
        int b = i >> 9; int k = i & 511;
        float v = (float)((const bf16*)A.pre_all)[((size_t)b * Lsteps + t) * Pdim + k];
        A.a_in[(size_t)b * IDIM + k] = v;
        A.p_in[(size_t)b * IDIM + k] = v;
      }
    }
    gsync(A.bar, A.gen);
  }
}

// ---------------------------------------------------------------------------
// Host
// ---------------------------------------------------------------------------
extern "C" void kernel_launch(void* const* d_in, const int* in_sizes, int n_in,
                              void* d_out, int out_size, void* d_ws, size_t ws_size,
                              hipStream_t stream) {
  (void)in_sizes; (void)n_in; (void)out_size; (void)ws_size;
  auto F = [&](int i) { return (const float*)d_in[i]; };
  // setup_inputs() flattening order (insertion order, depth-first):
  const float* enc_z = F(0);
  const float* enc_p = F(1);
  const float* ys    = F(2);
  const float* pre_w1 = F(3); const float* pre_b1 = F(4);
  const float* pre_w2 = F(5); const float* pre_b2 = F(6);
  const float *g_wih[4], *g_whh[4], *g_bih[4], *g_bhh[4];
  for (int g = 0; g < 4; ++g) { int base = 7 + g * 4;
    g_wih[g] = F(base); g_whh[g] = F(base + 1); g_bih[g] = F(base + 2); g_bhh[g] = F(base + 3); }
  const float *m_w1[2], *m_b1[2], *m_w2[2], *m_b2[2];
  for (int m = 0; m < 2; ++m) { int base = 23 + m * 4;
    m_w1[m] = F(base); m_b1[m] = F(base + 1); m_w2[m] = F(base + 2); m_b2[m] = F(base + 3); }
  const float *l_wih[2], *l_whh[2], *l_bih[2], *l_bhh[2];
  for (int l = 0; l < 2; ++l) { int base = 31 + l * 4;
    l_wih[l] = F(base); l_whh[l] = F(base + 1); l_bih[l] = F(base + 2); l_bhh[l] = F(base + 3); }
  const float* proj_w = F(39); const float* proj_b = F(40);
  const float* stop_w = F(41); const float* stop_b = F(42);

  char* base = (char*)d_ws;
  size_t off = 0;
  auto alloc = [&](size_t bytes) -> void* {
    void* p = base + off;
    off += (bytes + 255) & ~(size_t)255;
    return p;
  };

  // Control + recurrent state (zeroed every call for determinism; graph-capturable memset)
  size_t state_floats = 2 * 64 * 640 + 4 * 64 * 256 + 64 * 1280 + 2 * 2 * 64 * 256 + 64 * 512;
  size_t state_bytes = 256 + state_floats * 4;
  char* state = (char*)alloc(state_bytes);
  hipMemsetAsync(state, 0, state_bytes, stream);
  unsigned* bar = (unsigned*)state;
  unsigned* gen = bar + 1;
  float* a_in    = (float*)(state + 256);
  float* p_in    = a_in + 64 * 640;
  float* h_gru   = p_in + 64 * 640;
  float* res_in  = h_gru + 4 * 64 * 256;
  float* hl      = res_in + 64 * 1280;
  float* cl      = hl + 2 * 64 * 256;
  float* res_out = cl + 2 * 64 * 256;

  float* gi     = (float*)alloc((size_t)4 * 64 * 768 * 4);
  float* gh     = (float*)alloc((size_t)4 * 64 * 768 * 4);
  float* g_lstm = (float*)alloc((size_t)2 * 64 * 1024 * 4);
  float* q      = (float*)alloc(513 * 4);

  auto pack = [&](const float* W, int N, int K) -> unsigned short* {
    int ntiles = (N + 15) / 16, ktiles = (K + 31) / 32;
    long total = (long)ntiles * ktiles * 512;
    unsigned short* dst = (unsigned short*)alloc((size_t)total * 2);
    int blocks = (int)((total + 255) / 256);
    pack_w_kernel<<<blocks, 256, 0, stream>>>(W, dst, N, K, ktiles, total);
    return dst;
  };

  DecArgs A{};
  unsigned short* pk_pre1 = pack(pre_w1, 512, 80);
  unsigned short* pk_pre2 = pack(pre_w2, 512, 512);
  for (int g = 0; g < 4; ++g) {
    A.pk_gru_wih[g] = pack(g_wih[g], 768, 640);
    A.pk_gru_whh[g] = pack(g_whh[g], 768, 256);
    A.gru_bih[g] = g_bih[g]; A.gru_bhh[g] = g_bhh[g];
  }
  for (int l = 0; l < 2; ++l) {
    A.pk_lstm_wih[l] = pack(l_wih[l], 1024, 1280);
    A.pk_lstm_whh[l] = pack(l_whh[l], 1024, 256);
    A.lstm_bih[l] = l_bih[l]; A.lstm_bhh[l] = l_bhh[l];
  }
  A.pk_proj = pack(proj_w, 80, 512);

  size_t encN = (size_t)64 * 1024 * 128;
  unsigned short* enc_p_bf = (unsigned short*)alloc(encN * 2);
  unsigned short* enc_z_bf = (unsigned short*)alloc(encN * 2);
  unsigned short* pre_all  = (unsigned short*)alloc((size_t)64 * 500 * 512 * 2);
  {
    int blocks = (int)((encN + 255) / 256);
    f32_to_bf16_kernel<<<blocks, 256, 0, stream>>>(enc_p, enc_p_bf, (long)encN);
    f32_to_bf16_kernel<<<blocks, 256, 0, stream>>>(enc_z, enc_z_bf, (long)encN);
  }
  fuse_stop_kernel<<<1, 512, 0, stream>>>(proj_w, proj_b, stop_w, stop_b, q);
  prenet_kernel<<<2000, NTHR, 0, stream>>>(ys, pk_pre1, pre_b1, pk_pre2, pre_b2, pre_all);

  for (int m = 0; m < 2; ++m) { A.mol_w1[m] = m_w1[m]; A.mol_b1[m] = m_b1[m];
                                A.mol_w2[m] = m_w2[m]; A.mol_b2[m] = m_b2[m]; }
  A.proj_b = proj_b; A.q = q;
  A.enc_bf[0] = enc_p_bf;   // acoustic branch attends enc_p
  A.enc_bf[1] = enc_z_bf;   // prosody branch attends enc_z
  A.pre_all = pre_all;
  A.a_in = a_in; A.p_in = p_in; A.h_gru = h_gru; A.res_in = res_in;
  A.hl = hl; A.cl = cl; A.res_out = res_out;
  A.gi = gi; A.gh = gh; A.g_lstm = g_lstm;
  A.bar = bar; A.gen = gen;
  A.outs = (float*)d_out;
  A.logits = (float*)d_out + (size_t)64 * 500 * 80;

  decode_kernel<<<NWG, NTHR, 0, stream>>>(A);
}
